// AggregateSelfAttention_24790551232712
// MI455X (gfx1250) — matrix operand order repaired and checked by reading, not compiled, not run on hardware
//
#include <hip/hip_runtime.h>

// AggregateSelfAttention for MI455X (gfx1250, wave32, WMMA).
//
// Shapes: B=1, M=8192, D=768, C=2048, L=16.
// Dominant work: relu(vectors @ W1 + b1) -> 32768x768x768 GEMM = 38.7 GFLOP,
// f16 WMMA (f32 accumulate). All data is L2-resident (192 MB), so the design
// target is L2 bandwidth: each loaded W1 B-fragment is reused for G=4
// clusters' WMMAs in registers (4 clusters per workgroup), cutting packed-W1
// re-read traffic from 2.3 GB to ~0.6 GB, balancing it against ~8us of
// dense-f16 WMMA compute. `bout` is softmax-invariant and dropped.

typedef __attribute__((ext_vector_type(16))) _Float16 v16h;
typedef __attribute__((ext_vector_type(8)))  _Float16 v8h;
typedef __attribute__((ext_vector_type(8)))  float    v8f;

#define DDIM 768
#define CNUM 2048
#define LNUM 16
#define KT   (DDIM / 32)        // 24 K-steps of 32
#define NT   (DDIM / 16)        // 48 N-tiles of 16
#define NWAVES 8
#define NT_PER_WAVE (NT / NWAVES) // 6
#define JBLK 3                  // N-tiles per register block (6 = 2 x 3)
#define G    4                  // clusters per workgroup

// ---------------------------------------------------------------------------
// Pack W1 (f32 row-major [D][D]) into f16 WMMA B-fragments:
//   layout [kt][nt][lane(32)][elem(16)], 2 bytes each -> 1.13 MB in d_ws.
// B (32x16, 16-bit) per-lane layout: lane = hi*16 + n holds column n,
// K = (e<8 ? e : e+8) + hi*8 for elem e.
// ---------------------------------------------------------------------------
__global__ void __launch_bounds__(256)
pack_w1_kernel(const float* __restrict__ W1, _Float16* __restrict__ Wp) {
    int t = blockIdx.x * blockDim.x + threadIdx.x;   // < KT*NT*512
    int e    = t & 15;
    int lane = (t >> 4) & 31;
    int tile = t >> 9;          // kt*NT + nt
    int nt = tile % NT;
    int kt = tile / NT;
    if (kt >= KT) return;
    int n  = lane & 15;
    int hi = lane >> 4;
    int kk = (e < 8 ? e : e + 8) + hi * 8;       // K index within the 32-wide K block
    Wp[t] = (_Float16)W1[(size_t)(kt * 32 + kk) * DDIM + nt * 16 + n];
}

// ---------------------------------------------------------------------------
// One workgroup (8 wave32) per G=4 concept clusters.
// ---------------------------------------------------------------------------
__global__ void __launch_bounds__(256)
agg_attn_kernel(const float* __restrict__ mv,     // [M, D] f32
                const int*   __restrict__ cidx,   // [C, L]
                const int*   __restrict__ clen,   // [C]
                const _Float16* __restrict__ Wp,  // packed W1 fragments
                const float* __restrict__ b1,     // [D]
                const float* __restrict__ Wout,   // [D]
                float*       __restrict__ out)    // [C, D]
{
    // A fragments, per cluster: [kt][lane(32)][elem(16)] f16 = 24 KB; G = 96 KB.
    // A (16x32, 16-bit) per-lane layout: lane = hi*16 + m holds row m,
    // K = (e<8 ? e : e+8) + hi*8 for elem e.
    __shared__ alignas(32) _Float16 Apack[G * KT * 512];
    __shared__ float scores[G][LNUM];
    __shared__ float probs[G][LNUM];
    __shared__ int   rows[G][LNUM];

    const int c0  = blockIdx.x * G;
    const int tid = threadIdx.x;

    if (tid < G * LNUM) {
        int g = tid / LNUM, l = tid % LNUM;
        rows[g][l]   = cidx[(c0 + g) * LNUM + l];
        scores[g][l] = 0.0f;
    }
    __syncthreads();

    // Gather + convert the G*16 cluster rows into WMMA A-fragment order.
    // One chunk = 8 contiguous f32 (32 B, coalesced) -> one 16 B v8h LDS store.
    // chunk c2 of K-block kt: K = c2*8..c2*8+7 -> lane m + (c2&1)*16, elems (c2>>1)*8..+7.
    for (int t = tid; t < G * LNUM * KT * 4; t += 256) {
        int c2 = t & 3;
        int kt = (t >> 2) % KT;
        int m  = ((t >> 2) / KT) & 15;
        int g  = t >> (2 + 5 + 4);          // / (4*KT*16) with KT=24 -> use div
        g = t / (4 * KT * LNUM);
        const float4* p =
            (const float4*)(mv + (size_t)rows[g][m] * DDIM + kt * 32 + c2 * 8);
        float4 x = p[0], y = p[1];
        v8h h;
        h[0] = (_Float16)x.x; h[1] = (_Float16)x.y;
        h[2] = (_Float16)x.z; h[3] = (_Float16)x.w;
        h[4] = (_Float16)y.x; h[5] = (_Float16)y.y;
        h[6] = (_Float16)y.z; h[7] = (_Float16)y.w;
        int lane  = m + (c2 & 1) * 16;
        int ebase = (c2 >> 1) * 8;
        *(v8h*)(Apack + (size_t)(g * KT + kt) * 512 + lane * 16 + ebase) = h;
    }
    __syncthreads();

    // ---- GEMM: h = relu(vecs @ W1 + b1), fused with scores = h . Wout ----
    const int wave   = tid >> 5;
    const int lane   = tid & 31;
    const int ntbase = wave * NT_PER_WAVE;
    const int n  = lane & 15;
    const int hi = lane >> 4;

    float loc[G][8];                      // score partials, persist across blocks
#pragma unroll
    for (int g = 0; g < G; ++g)
#pragma unroll
        for (int r = 0; r < 8; ++r) loc[g][r] = 0.0f;

    for (int jb = 0; jb < NT_PER_WAVE / JBLK; ++jb) {
        const int nt0 = ntbase + jb * JBLK;
        v8f acc[G][JBLK];
#pragma unroll
        for (int g = 0; g < G; ++g)
#pragma unroll
            for (int j = 0; j < JBLK; ++j) acc[g][j] = (v8f){};

        for (int kt = 0; kt < KT; ++kt) {
            v16h a[G];
#pragma unroll
            for (int g = 0; g < G; ++g)
                a[g] = *(const v16h*)(Apack + (size_t)(g * KT + kt) * 512 + lane * 16);
            const _Float16* bp = Wp + (size_t)(kt * NT + nt0) * 512 + lane * 16;
#pragma unroll
            for (int j = 0; j < JBLK; ++j) {
                v16h b = *(const v16h*)(bp + j * 512);   // reused G times
#pragma unroll
                for (int g = 0; g < G; ++g)
                    acc[g][j] = __builtin_amdgcn_wmma_f32_16x16x32_f16(
                        false, a[g], false, b, (short)0, acc[g][j], false, false);
            }
        }

        // Fold bias + relu + Wout dot into score partials before acc is reused.
        // C/D layout: lane = hi*16 + n is column n; reg r is row m = hi*8 + r.
#pragma unroll
        for (int j = 0; j < JBLK; ++j) {
            int col = (nt0 + j) * 16 + n;
            float bias = b1[col];
            float wo   = Wout[col];
#pragma unroll
            for (int g = 0; g < G; ++g)
#pragma unroll
                for (int r = 0; r < 8; ++r) {
                    float h = acc[g][j][r] + bias;
                    h = h > 0.0f ? h : 0.0f;     // relu
                    loc[g][r] += h * wo;
                }
        }
    }

    // Reduce across the 16 lanes sharing each row: ds_swizzle xor 1,2,4,8.
#pragma unroll
    for (int g = 0; g < G; ++g)
#pragma unroll
        for (int r = 0; r < 8; ++r) {
            float v = loc[g][r];
            v += __int_as_float(__builtin_amdgcn_ds_swizzle(__float_as_int(v), 0x041f));
            v += __int_as_float(__builtin_amdgcn_ds_swizzle(__float_as_int(v), 0x081f));
            v += __int_as_float(__builtin_amdgcn_ds_swizzle(__float_as_int(v), 0x101f));
            v += __int_as_float(__builtin_amdgcn_ds_swizzle(__float_as_int(v), 0x201f));
            loc[g][r] = v;
        }
    if (n == 0) {
#pragma unroll
        for (int g = 0; g < G; ++g)
#pragma unroll
            for (int r = 0; r < 8; ++r)
                atomicAdd(&scores[g][hi * 8 + r], loc[g][r]);   // ds_add_f32
    }
    __syncthreads();

    // ---- Masked softmax over L=16 per cluster (bout cancels) ----
    if (tid < G) {
        int g = tid;
        int len = clen[c0 + g];
        len = len < 1 ? 1 : (len > LNUM ? LNUM : len);
        float mx = -3.4e38f;
        for (int l = 0; l < len; ++l) mx = fmaxf(mx, scores[g][l]);
        float s = 0.0f, e[LNUM];
#pragma unroll
        for (int l = 0; l < LNUM; ++l) {
            float v = (l < len) ? __expf(scores[g][l] - mx) : 0.0f;
            e[l] = v; s += v;
        }
        float inv = 1.0f / s;
#pragma unroll
        for (int l = 0; l < LNUM; ++l) probs[g][l] = e[l] * inv;
    }
    __syncthreads();

    // ---- output[c, :] = probs . vectors (f32 rows re-read from L2) ----
    for (int t = tid; t < G * DDIM; t += 256) {
        int g  = t / DDIM;
        int d0 = t % DDIM;
        float s = 0.0f;
#pragma unroll
        for (int l = 0; l < LNUM; ++l)
            s += probs[g][l] * mv[(size_t)rows[g][l] * DDIM + d0];
        out[(size_t)(c0 + g) * DDIM + d0] = s;
    }
}

extern "C" void kernel_launch(void* const* d_in, const int* in_sizes, int n_in,
                              void* d_out, int out_size, void* d_ws, size_t ws_size,
                              hipStream_t stream) {
    const float* mv   = (const float*)d_in[0];  // mention_vectors [1,8192,768] f32
    const int*   cidx = (const int*)d_in[1];    // concept_indices [2048,16]
    const int*   clen = (const int*)d_in[2];    // concept_lengths [2048]
    const float* W1   = (const float*)d_in[3];  // [768,768] f32
    const float* b1   = (const float*)d_in[4];  // [768] f32
    const float* Wout = (const float*)d_in[5];  // [768] f32
    // d_in[6] = bout: softmax-invariant, unused.

    _Float16* Wp = (_Float16*)d_ws;             // 1.13 MB packed W1 fragments

    const int packElems = KT * NT * 512;        // 589824
    pack_w1_kernel<<<packElems / 256, 256, 0, stream>>>(W1, Wp);
    agg_attn_kernel<<<CNUM / G, 256, 0, stream>>>(mv, cidx, clen, Wp, b1, Wout,
                                                  (float*)d_out);
}
